// Encoder_23708219474392
// MI455X (gfx1250) — matrix operand order repaired and checked by reading, not compiled, not run on hardware
//
#include <hip/hip_runtime.h>

// ---------------------------------------------------------------------------
// LSTM encoder for MI455X (gfx1250, wave32, WMMA + TDM).
//   Phase 1: x_proj = inputs @ Wx + b  (bf16 WMMA, fp32 accum), stored in
//            register-image tile layout (tile*256 + lane*8) for phase 2.
//   Phase 2: persistent kernel, 1024 recurrent steps with grid-wide barrier;
//            Wh slice + per-step h tile staged to LDS via tensor_load_to_lds;
//            cell state c resident in registers for all 1024 steps.
// ---------------------------------------------------------------------------

typedef float  v8f   __attribute__((ext_vector_type(8)));
typedef __bf16 v16bf __attribute__((ext_vector_type(16)));
typedef __bf16 v8bf  __attribute__((ext_vector_type(8)));
typedef unsigned int u32x4 __attribute__((ext_vector_type(4)));
typedef int          i32x4 __attribute__((ext_vector_type(4)));
typedef int          i32x8 __attribute__((ext_vector_type(8)));

static constexpr int B_ = 64;
static constexpr int T_ = 1024;
static constexpr int F_ = 512;
static constexpr int H_ = 512;
static constexpr int G_ = 4 * H_;      // 2048 gate columns
static constexpr int M_ = B_ * T_;     // 65536 rows of phase-1 GEMM
static constexpr int NT_ = G_ / 16;    // 128 n-tiles per row block

#define DEV_INLINE __device__ __forceinline__

DEV_INLINE float sigf(float x)      { return 1.0f / (1.0f + __expf(-x)); }
DEV_INLINE float tanh_fast(float x) { return 1.0f - 2.0f / (1.0f + __expf(2.0f * x)); }

DEV_INLINE v16bf cat8(v8bf lo, v8bf hi) {
  return __builtin_shufflevector(lo, hi, 0,1,2,3,4,5,6,7,8,9,10,11,12,13,14,15);
}

// A-operand tile (16x32 bf16), K-major rows, leading dim ld.
// Lane holds row M=(lane&15); K-halves: lanes<16 -> K {0..15},
// lanes>=16 -> K {16..31}.
DEV_INLINE v16bf load_a(const __bf16* __restrict__ base, int ld, int r, int half, int k0) {
  const __bf16* p = base + (size_t)r * ld + k0 + half * 16;
  v8bf lo = *(const v8bf*)(p);
  v8bf hi = *(const v8bf*)(p + 8);
  return cat8(lo, hi);
}

// B-operand tile (32x16 bf16) from N-major rows of K: lane -> row N=(lane&15),
// 16 contiguous K values, K-half selected by lane>>4.
DEV_INLINE v16bf load_b(const __bf16* __restrict__ base, int ld, int r, int half, int k0) {
  return *(const v16bf*)(base + (size_t)r * ld + k0 + half * 16);
}

// ---- Tensor Data Mover: 2-D bf16 tile Global -> LDS (ISA ch.8 D# layout) ---
#if __has_builtin(__builtin_amdgcn_tensor_load_to_lds)
#define HAVE_TDM 1
DEV_INLINE void tdm_load_bf16_2d(unsigned lds_byte_off, const void* gaddr,
                                 unsigned rows, unsigned kelems) {
  const unsigned long long ga = (unsigned long long)gaddr;
  u32x4 g0;
  g0[0] = 1u;                                   // count=1, user descriptor
  g0[1] = lds_byte_off;                         // LDS byte address of tile
  g0[2] = (unsigned)(ga & 0xffffffffu);         // global_addr[31:0]
  g0[3] = (unsigned)((ga >> 32) & 0x01ffffffu)  // global_addr[56:32]
          | (2u << 30);                         // type = 2 ("image")
  i32x8 g1;
  g1[0] = (int)(1u << 16);                      // wg_mask=0, data_size=1 (2B)
  g1[1] = (int)((kelems & 0xffffu) << 16);      // tensor_dim0[15:0] @ bits 63:48
  g1[2] = (int)((kelems >> 16) | ((rows & 0xffffu) << 16)); // dim0 hi | dim1 lo
  g1[3] = (int)((rows >> 16) | (kelems << 16)); // dim1 hi | tile_dim0
  g1[4] = (int)(rows & 0xffffu);                // tile_dim1 | tile_dim2=0
  g1[5] = (int)kelems;                          // tensor_dim0_stride[31:0]
  g1[6] = 0;                                    // stride0 hi | dim1_stride lo
  g1[7] = 0;
  const i32x4 gz4 = {0, 0, 0, 0};               // groups 2/3 unused (2-D tile)
  const i32x8 gz8 = {0, 0, 0, 0, 0, 0, 0, 0};   // extra group (clang-23 form)
  // cpol: TH[2:0]=0 (RT), SCOPE[4:3]=2 (DEV) so reloads miss stale WGP$.
  __builtin_amdgcn_tensor_load_to_lds(g0, g1, gz4, gz4, gz8, (2 << 3));
}
#else
#define HAVE_TDM 0
#endif

// ------------------------- prep kernels ------------------------------------

// inputs [B,T,F] fp32 -> xb [(t*B+b), F] bf16  (row remap for phase-1 A)
__global__ void k_cvt_inputs(const float* __restrict__ in, __bf16* __restrict__ xb) {
  const long long N = (long long)B_ * T_ * F_;
  for (long long i = blockIdx.x * (long long)blockDim.x + threadIdx.x; i < N;
       i += (long long)gridDim.x * blockDim.x) {
    int f = (int)(i % F_);
    long long bt = i / F_;
    int t = (int)(bt % T_);
    int b = (int)(bt / T_);
    xb[((long long)t * B_ + b) * F_ + f] = (__bf16)in[i];
  }
}

// W [K, G] fp32 -> Wt [G, K] bf16 (transpose+convert), K = 512 for Wx and Wh
__global__ void k_cvt_w(const float* __restrict__ w, __bf16* __restrict__ wt, int K) {
  const long long N = (long long)K * G_;
  for (long long i = blockIdx.x * (long long)blockDim.x + threadIdx.x; i < N;
       i += (long long)gridDim.x * blockDim.x) {
    int g = (int)(i % G_);
    int k = (int)(i / G_);
    wt[(long long)g * K + k] = (__bf16)w[i];
  }
}

__global__ void k_init(__bf16* __restrict__ hb, unsigned* __restrict__ cnt) {
  int i = blockIdx.x * blockDim.x + threadIdx.x;
  if (i < 2 * B_ * H_) hb[i] = (__bf16)0.0f;   // zero both h double-buffers
  if (i == 0) *cnt = 0u;                       // grid-barrier counter
}

// ------------------------- phase 1: x-projection GEMM ----------------------
// Output layout: x_proj tile (mtGlobal, nt) stored as 256 contiguous floats at
// (mtGlobal*NT_ + nt)*256 + lane*8  (the WMMA C register image, v8f per lane).

__global__ __launch_bounds__(256) void
k_xproj(const __bf16* __restrict__ xb, const __bf16* __restrict__ wxT,
        const float* __restrict__ bias, float* __restrict__ xproj) {
  const int lane = threadIdx.x & 31;
  const int wave = threadIdx.x >> 5;
  const int half = lane >> 4;
  const int r    = lane & 15;
  const int nWaves = (gridDim.x * blockDim.x) >> 5;
  const int wgl = blockIdx.x * (blockDim.x >> 5) + wave;
  const int MT = M_ / 16;

  for (int tile = wgl; tile < MT * NT_; tile += nWaves) {
    const int mt = tile / NT_, nt = tile % NT_;
    const int m0 = mt * 16, n0 = nt * 16;
    const float bv = bias[n0 + r];
    v8f acc = {bv, bv, bv, bv, bv, bv, bv, bv};
    const __bf16* arow = xb + (size_t)m0 * F_;
#pragma unroll
    for (int k0 = 0; k0 < F_; k0 += 32) {
      v16bf a = load_a(arow, F_, r, half, k0);
      v16bf b = load_b(wxT + (size_t)n0 * F_, F_, r, half, k0);
      acc = __builtin_amdgcn_wmma_f32_16x16x32_bf16(false, a, false, b,
                                                    (short)0, acc, false, false);
    }
    *(v8f*)(xproj + (size_t)tile * 256 + lane * 8) = acc;   // coalesced 32B/lane
  }
}

// ------------------------- phase 2: recurrent scan -------------------------
// 32 blocks x 128 threads (4 waves). Block covers one m-tile (mt = blk>>3)
// and 4 consecutive j-tiles (jt = (blk&7)*4 + wave). Each wave computes the
// four gate tiles of its (m,j) tile; c stays in registers for all steps.

static constexpr int SWH_ELEMS  = 4 * 64 * 512;           // Wh^T slice: 256 KB
static constexpr int LDSA_ELEMS = 16 * 512;               // h tile: 16 KB
static constexpr unsigned LDSA_BYTE_OFF = SWH_ELEMS * 2;  // 262144
static constexpr size_t LSTM_LDS_BYTES = (size_t)(SWH_ELEMS + LDSA_ELEMS) * 2;

__global__ __launch_bounds__(128) void
k_lstm(const float* __restrict__ xproj, const __bf16* __restrict__ whT,
       __bf16* __restrict__ hb, float* __restrict__ out,
       unsigned* __restrict__ cnt, int nBlocks) {
  extern __shared__ __bf16 smem[];
  __bf16* swh  = smem;               // [4 gates][64 rows][512 K]
  __bf16* ldsa = smem + SWH_ELEMS;   // [16 rows][512 K]

  const int lane = threadIdx.x & 31;
  const int wave = threadIdx.x >> 5;         // 0..3
  const int half = lane >> 4;
  const int r    = lane & 15;
  const int mt = blockIdx.x >> 3;            // 0..3
  const int jb = (blockIdx.x & 7) * 4;       // block's first j-tile
  const int jt = jb + wave;                  // this wave's j-tile (0..31)
  const int m0 = mt * 16;
  const int j0 = jt * 16;

  // ---- stage this block's Wh^T slice into LDS (once) ----
  // gate g rows: n = g*512 + jb*16 + nL, nL in [0,64); wave w DMAs gate w.
#if HAVE_TDM
  {
    const int g = wave;                      // one 64x512 slab per wave
    tdm_load_bf16_2d((unsigned)(g * 64 * 512 * 2),
                     whT + ((size_t)g * 512 + jb * 16) * 512,
                     /*rows=*/64, /*kelems=*/512);
    __builtin_amdgcn_s_wait_tensorcnt(0);
  }
#else
  for (int ch = threadIdx.x; ch < 256 * 64; ch += blockDim.x) {
    const int row = ch >> 6;                 // g*64 + nL
    const int kc  = ch & 63;                 // 16-byte chunk in row
    const int g = row >> 6, nL = row & 63;
    const int n = g * 512 + jb * 16 + nL;
    *(uint4*)(swh + (size_t)row * 512 + kc * 8) =
        *(const uint4*)(whT + (size_t)n * 512 + kc * 8);
  }
#endif
  __syncthreads();

  v8f cacc  = {0.f, 0.f, 0.f, 0.f, 0.f, 0.f, 0.f, 0.f};
  v8f hlast = {0.f, 0.f, 0.f, 0.f, 0.f, 0.f, 0.f, 0.f};

  for (int t = 0; t < T_; ++t) {
    const __bf16* hin  = hb + (size_t)(t & 1) * (B_ * H_);
    __bf16*       hout = hb + (size_t)((t + 1) & 1) * (B_ * H_);

    // ---- stage the block's h tile (rows m0..m0+15, all K) into LDS ----
#if HAVE_TDM
    if (wave == 0) {
      tdm_load_bf16_2d(LDSA_BYTE_OFF, hin + (size_t)m0 * H_,
                       /*rows=*/16, /*kelems=*/512);
    }
#else
    for (int ch = threadIdx.x; ch < 16 * 64; ch += blockDim.x) {
      const int rr = ch >> 6, kc = ch & 63;
      *(uint4*)(ldsa + (size_t)rr * 512 + kc * 8) =
          *(const uint4*)(hin + (size_t)(m0 + rr) * H_ + kc * 8);
    }
#endif

    // ---- acc init = x_proj tile per gate (register-image layout) ----
    // overlaps with the TDM transfer above
    v8f ai, af, ag, ao;
    {
      const size_t trow = (size_t)(t * 4 + mt) * NT_;
      ai = *(const v8f*)(xproj + (trow + 0 * 32 + jt) * 256 + lane * 8);
      af = *(const v8f*)(xproj + (trow + 1 * 32 + jt) * 256 + lane * 8);
      ag = *(const v8f*)(xproj + (trow + 2 * 32 + jt) * 256 + lane * 8);
      ao = *(const v8f*)(xproj + (trow + 3 * 32 + jt) * 256 + lane * 8);
    }

#if HAVE_TDM
    if (wave == 0) __builtin_amdgcn_s_wait_tensorcnt(0);
#endif
    __syncthreads();

    // ---- gates += h @ Wh : one A tile feeds 4 WMMAs per K-step ----
#pragma unroll
    for (int k0 = 0; k0 < H_; k0 += 32) {
      v16bf a  = load_a(ldsa, 512, r, half, k0);
      v16bf b0 = load_b(swh + (size_t)(0 * 64 + wave * 16) * 512, 512, r, half, k0);
      v16bf b1 = load_b(swh + (size_t)(1 * 64 + wave * 16) * 512, 512, r, half, k0);
      v16bf b2 = load_b(swh + (size_t)(2 * 64 + wave * 16) * 512, 512, r, half, k0);
      v16bf b3 = load_b(swh + (size_t)(3 * 64 + wave * 16) * 512, 512, r, half, k0);
      ai = __builtin_amdgcn_wmma_f32_16x16x32_bf16(false, a, false, b0, (short)0, ai, false, false);
      af = __builtin_amdgcn_wmma_f32_16x16x32_bf16(false, a, false, b1, (short)0, af, false, false);
      ag = __builtin_amdgcn_wmma_f32_16x16x32_bf16(false, a, false, b2, (short)0, ag, false, false);
      ao = __builtin_amdgcn_wmma_f32_16x16x32_bf16(false, a, false, b3, (short)0, ao, false, false);
    }

    // ---- LSTM cell (fp32); h re-quantized to bf16 for next step's A ----
#pragma unroll
    for (int v = 0; v < 8; ++v) {
      const float iv = sigf(ai[v]);
      const float fv = sigf(af[v]);
      const float gv = tanh_fast(ag[v]);
      const float ov = sigf(ao[v]);
      const float cv = fv * cacc[v] + iv * gv;
      cacc[v] = cv;
      const float hv = ov * tanh_fast(cv);
      hlast[v] = hv;
      hout[(size_t)(m0 + v + half * 8) * H_ + j0 + r] = (__bf16)hv;
    }

    // ---- grid-wide barrier (release/acquire, agent scope) ----
    __syncthreads();
    if (threadIdx.x == 0) {
      __hip_atomic_fetch_add(cnt, 1u, __ATOMIC_RELEASE, __HIP_MEMORY_SCOPE_AGENT);
      const unsigned target = (unsigned)nBlocks * (unsigned)(t + 1);
      while (__hip_atomic_load(cnt, __ATOMIC_ACQUIRE, __HIP_MEMORY_SCOPE_AGENT) < target)
        __builtin_amdgcn_s_sleep(1);
    }
    __syncthreads();
  }

  // final carry: out[0:BH] = c, out[BH:2BH] = h (fp32, straight from registers)
#pragma unroll
  for (int v = 0; v < 8; ++v) {
    const int row = m0 + v + half * 8;
    const int col = j0 + r;
    out[(size_t)row * H_ + col] = cacc[v];
    out[(size_t)(B_ * H_) + (size_t)row * H_ + col] = hlast[v];
  }
}

// ------------------------- host side ---------------------------------------

extern "C" void kernel_launch(void* const* d_in, const int* in_sizes, int n_in,
                              void* d_out, int out_size, void* d_ws, size_t ws_size,
                              hipStream_t stream) {
  (void)in_sizes; (void)n_in; (void)out_size; (void)ws_size;
  const float* in   = (const float*)d_in[0];   // [B,T,F]
  const float* Wx   = (const float*)d_in[1];   // [F,4H]
  const float* Wh   = (const float*)d_in[2];   // [H,4H]
  const float* bias = (const float*)d_in[3];   // [4H]
  float* out = (float*)d_out;                  // c[B,H] ++ h[B,H]

  char* ws = (char*)d_ws;
  size_t o = 0;
  float*  xproj = (float*)(ws + o);  o += (size_t)T_ * B_ * G_ * sizeof(float);   // 512 MB
  __bf16* xb    = (__bf16*)(ws + o); o += (size_t)M_ * F_ * 2;                    //  64 MB
  __bf16* wxT   = (__bf16*)(ws + o); o += (size_t)G_ * F_ * 2;                    //   2 MB
  __bf16* whT   = (__bf16*)(ws + o); o += (size_t)G_ * H_ * 2;                    //   2 MB
  __bf16* hb    = (__bf16*)(ws + o); o += (size_t)2 * B_ * H_ * 2;                // 128 KB
  unsigned* cnt = (unsigned*)(ws + o);

  k_cvt_inputs<<<8192, 256, 0, stream>>>(in, xb);
  k_cvt_w<<<2048, 256, 0, stream>>>(Wx, wxT, F_);
  k_cvt_w<<<2048, 256, 0, stream>>>(Wh, whT, H_);
  k_init<<<(2 * B_ * H_ + 255) / 256, 256, 0, stream>>>(hb, cnt);

  k_xproj<<<1024, 256, 0, stream>>>(xb, wxT, bias, xproj);

  const int nBlocks = 32;
  k_lstm<<<nBlocks, 128, LSTM_LDS_BYTES, stream>>>(xproj, whT, hb, out, cnt, nBlocks);
}
